// XLSTM_55310588838109
// MI455X (gfx1250) — matrix operand order repaired and checked by reading, not compiled, not run on hardware
//
#include <hip/hip_runtime.h>
#include <hip/hip_bf16.h>

// ---------------- problem constants ----------------
constexpr int Bsz   = 16;
constexpr int Csz   = 862;     // time length for the xLSTM part
constexpr int Lsz   = 336;
constexpr int PRED  = 96;
constexpr int N2    = 256;
constexpr int INNER = 512;
constexpr int NHh   = 4;
constexpr int DHh   = 128;
constexpr int Mrows = Bsz * Csz;    // 13792
constexpr int Mp    = 13824;        // 216 * 64 (padded M for 64-row GEMM tiles)
constexpr int Tp    = 896;          // 14 * 64 (padded T for attention tiles)
constexpr int T_SEQ = 862;

typedef __attribute__((ext_vector_type(16))) _Float16 v16h;
typedef __attribute__((ext_vector_type(8)))  _Float16 v8h;
typedef __attribute__((ext_vector_type(8)))  float    v8f;

// ---------------- WMMA helpers ----------------
__device__ inline v8f wmma_f16(v16h a, v16h b, v8f c) {
  return __builtin_amdgcn_wmma_f32_16x16x32_f16(false, a, false, b, (short)0, c,
                                                false, false);
}

// A fragment: 16x32 f16, source row-major (ld in halves).
// lane<16: K = {k0..k0+7, k0+16..k0+23}; lane>=16: K = {k0+8..15, k0+24..31}.
__device__ inline v16h load_a(const _Float16* base, int ld, int row0, int k0, int lane) {
  int r  = row0 + (lane & 15);
  int hi = (lane >> 4) * 8;
  const _Float16* p = base + (size_t)r * ld + k0 + hi;
  v8h lo = *(const v8h*)p;
  v8h hh = *(const v8h*)(p + 16);
  v16h out;
#pragma unroll
  for (int i = 0; i < 8; ++i) { out[i] = lo[i]; out[i + 8] = hh[i]; }
  return out;
}

// B fragment: 32x16 (KxN) f16; source is Bt = N x K row-major (column n of B
// contiguous over K). lane<16: K = {k0..k0+7, k0+16..k0+23}; lane>=16: K = {k0+8..15, k0+24..31}.
__device__ inline v16h load_b(const _Float16* base, int ld, int col0, int k0, int lane) {
  int c  = col0 + (lane & 15);
  int hi = (lane >> 4) * 8;
  const _Float16* p = base + (size_t)c * ld + k0 + hi;
  v8h lo = *(const v8h*)p;
  v8h hh = *(const v8h*)(p + 16);
  v16h out;
#pragma unroll
  for (int i = 0; i < 8; ++i) { out[i] = lo[i]; out[i + 8] = hh[i]; }
  return out;
}

// ---------------- generic GEMM: C = A @ B^T (+bias)(+res) ----------------
// A: (Mp x lda) f16 row-major (rows >= M zero-padded)
// Bt:(Nb x ldb) f16 row-major (rows >= N zero-padded)
// block 128 thr = 4 waves (2x2), wave tile 32x32, block tile 64x64.
__global__ __launch_bounds__(128)
void k_gemm(const _Float16* __restrict__ A, const _Float16* __restrict__ Bt,
            float* __restrict__ C, _Float16* __restrict__ C16,
            const float* __restrict__ bias, const float* __restrict__ res,
            int M, int N, int K, int lda, int ldb, int ldc, int ldc16) {
  const int lane = threadIdx.x & 31;
  const int wave = threadIdx.x >> 5;
  const int m0 = blockIdx.y * 64 + (wave >> 1) * 32;
  const int n0 = blockIdx.x * 64 + (wave & 1) * 32;
  v8f a00 = {}, a01 = {}, a10 = {}, a11 = {};
  for (int k0 = 0; k0 < K; k0 += 32) {
    v16h fa0 = load_a(A, lda, m0,      k0, lane);
    v16h fa1 = load_a(A, lda, m0 + 16, k0, lane);
    v16h fb0 = load_b(Bt, ldb, n0,      k0, lane);
    v16h fb1 = load_b(Bt, ldb, n0 + 16, k0, lane);
    a00 = wmma_f16(fa0, fb0, a00);
    a01 = wmma_f16(fa0, fb1, a01);
    a10 = wmma_f16(fa1, fb0, a10);
    a11 = wmma_f16(fa1, fb1, a11);
  }
  const int nc  = lane & 15;
  const int rof = 8 * (lane >> 4);
#pragma unroll
  for (int tm = 0; tm < 2; ++tm)
#pragma unroll
    for (int tn = 0; tn < 2; ++tn) {
      v8f acc = tm == 0 ? (tn == 0 ? a00 : a01) : (tn == 0 ? a10 : a11);
      int n = n0 + tn * 16 + nc;
      if (n >= N) continue;
      float bv = bias ? bias[n] : 0.f;
#pragma unroll
      for (int r = 0; r < 8; ++r) {
        int row = m0 + tm * 16 + rof + r;
        if (row < M) {
          float v = acc[r] + bv;
          if (res) v += res[(size_t)row * ldc + n];
          if (C)   C[(size_t)row * ldc + n] = v;
          if (C16) C16[(size_t)row * ldc16 + n] = (_Float16)v;
        }
      }
    }
}

// ---------------- small helpers ----------------
__device__ inline float gelu_exact(float x) {
  return 0.5f * x * (1.f + erff(x * 0.70710678118654752f));
}
__device__ inline float silu_f(float x) { return x / (1.f + __expf(-x)); }

__global__ void k_zero_h(_Float16* p, size_t n) {
  size_t i = (size_t)blockIdx.x * 256 + threadIdx.x;
  if (i < n) p[i] = (_Float16)0.f;
}

// trend = 25-tap edge-padded moving average along L (contiguous axis)
__global__ __launch_bounds__(352)
void k_trend(const float* __restrict__ x, float* __restrict__ trend) {
  int row = blockIdx.x, l = threadIdx.x;
  if (l >= Lsz) return;
  const float* xr = x + (size_t)row * Lsz;
  float s = 0.f;
  for (int j = -12; j <= 12; ++j) {
    int p = l + j; p = p < 0 ? 0 : (p > Lsz - 1 ? Lsz - 1 : p);
    s += xr[p];
  }
  trend[(size_t)row * Lsz + l] = s * (1.f / 25.f);
}

// A1 = [x - trend | trend] packed f16, (Mp x 672), zero pad rows
__global__ void k_packA1(const float* __restrict__ x, const float* __restrict__ tr,
                         _Float16* __restrict__ A1) {
  size_t idx = (size_t)blockIdx.x * 256 + threadIdx.x;
  if (idx >= (size_t)Mp * 672) return;
  int m = (int)(idx / 672), k = (int)(idx % 672);
  float v = 0.f;
  if (m < Mrows) {
    if (k < Lsz) v = x[(size_t)m * Lsz + k] - tr[(size_t)m * Lsz + k];
    else         v = tr[(size_t)m * Lsz + (k - Lsz)];
  }
  A1[idx] = (_Float16)v;
}

// W1 = [Wseas | Wtrend] packed f16, (384 x 672), zero pad rows
__global__ void k_packW1(const float* __restrict__ Ws, const float* __restrict__ Wt,
                         _Float16* __restrict__ W1) {
  size_t idx = (size_t)blockIdx.x * 256 + threadIdx.x;
  if (idx >= (size_t)384 * 672) return;
  int n = (int)(idx / 672), k = (int)(idx % 672);
  float v = 0.f;
  if (n < Lsz) v = (k < Lsz) ? Ws[(size_t)n * Lsz + k] : Wt[(size_t)n * Lsz + (k - Lsz)];
  W1[idx] = (_Float16)v;
}

__global__ void k_bsum(const float* a, const float* b, float* o) {
  int i = blockIdx.x * 256 + threadIdx.x;
  if (i < Lsz) o[i] = a[i] + b[i];
}

// generic f32 (N x K) -> f16 (Nb x Kp) padded convert
__global__ void k_cvtW(const float* __restrict__ src, _Float16* __restrict__ dst,
                       int N, int K, int Kp, int Nb) {
  size_t idx = (size_t)blockIdx.x * 256 + threadIdx.x;
  if (idx >= (size_t)Nb * Kp) return;
  int n = (int)(idx / Kp), k = (int)(idx % Kp);
  dst[idx] = (n < N && k < K) ? (_Float16)src[(size_t)n * K + k] : (_Float16)0.f;
}

// BatchNorm statistics over (B, N2) per channel c
__global__ __launch_bounds__(256)
void k_bnstats(const float* __restrict__ h, float* mu, float* rstd) {
  int c = blockIdx.x;
  float s = 0.f, s2 = 0.f;
  for (int bb = 0; bb < Bsz; ++bb) {
    float v = h[((size_t)(bb * Csz + c)) * N2 + threadIdx.x];
    s += v; s2 += v * v;
  }
  __shared__ float sm[256], sq[256];
  sm[threadIdx.x] = s; sq[threadIdx.x] = s2; __syncthreads();
  for (int st = 128; st > 0; st >>= 1) {
    if (threadIdx.x < st) { sm[threadIdx.x] += sm[threadIdx.x + st]; sq[threadIdx.x] += sq[threadIdx.x + st]; }
    __syncthreads();
  }
  if (threadIdx.x == 0) {
    float m = sm[0] / 4096.f;
    mu[c] = m;
    rstd[c] = rsqrtf(sq[0] / 4096.f - m * m + 1e-5f);
  }
}

// fused: BN apply + GELU (-> residual h0) + LayerNorm(ln1_w) -> f16
__global__ __launch_bounds__(256)
void k_ln1(const float* __restrict__ h, const float* mu, const float* rstd,
           const float* bn_g, const float* bn_b, const float* ln1w,
           float* __restrict__ h0, _Float16* __restrict__ xnh) {
  int row = blockIdx.x, n = threadIdx.x;
  int c = row % Csz;
  float v  = h[(size_t)row * N2 + n];
  float bn = (v - mu[c]) * rstd[c] * bn_g[c] + bn_b[c];
  float gl = gelu_exact(bn);
  h0[(size_t)row * N2 + n] = gl;
  __shared__ float sm[256], sq[256];
  sm[n] = gl; sq[n] = gl * gl; __syncthreads();
  for (int st = 128; st > 0; st >>= 1) {
    if (n < st) { sm[n] += sm[n + st]; sq[n] += sq[n + st]; }
    __syncthreads();
  }
  float m   = sm[0] / (float)N2;
  float var = sq[0] / (float)N2 - m * m;
  xnh[(size_t)row * N2 + n] = (_Float16)((gl - m) * rsqrtf(var + 1e-5f) * ln1w[n]);
}

// depthwise causal conv (k=4) over time + SiLU; x_m = xi[:, :512]
__global__ __launch_bounds__(512)
void k_conv(const float* __restrict__ xi, const float* __restrict__ cw,
            const float* __restrict__ cb, float* __restrict__ xa) {
  int row = blockIdx.x, i = threadIdx.x;
  int t = row % Csz;
  float acc = cb[i];
#pragma unroll
  for (int kk = 0; kk < 4; ++kk) {
    int dt = kk - 3;
    if (t + dt >= 0) acc += cw[i * 4 + kk] * xi[(size_t)(row + dt) * 1024 + i];
  }
  xa[(size_t)row * INNER + i] = silu_f(acc);
}

// block-diagonal (128 x 4x4) projections; write f16 (B,NH,Tp,DH) operands
__global__ __launch_bounds__(512)
void k_headwise(const float* __restrict__ xa, const float* __restrict__ xi,
                const float* __restrict__ Wq, const float* __restrict__ Wk,
                const float* __restrict__ Wv,
                _Float16* __restrict__ qh, _Float16* __restrict__ kh,
                _Float16* __restrict__ vh) {
  int row = blockIdx.x, ch = threadIdx.x;
  int b = row / Csz, t = row % Csz;
  int gI = ch >> 2, o = ch & 3;
  float qv = 0.f, kv = 0.f, vv = 0.f;
#pragma unroll
  for (int i = 0; i < 4; ++i) {
    float a = xa[(size_t)row * INNER + gI * 4 + i];
    float m = xi[(size_t)row * 1024 + gI * 4 + i];
    float wqi = Wq[gI * 16 + o * 4 + i], wki = Wk[gI * 16 + o * 4 + i], wvi = Wv[gI * 16 + o * 4 + i];
    qv += a * wqi; kv += a * wki; vv += m * wvi;
  }
  int hd = ch >> 7, d = ch & 127;
  size_t idx = (((size_t)(b * NHh + hd)) * Tp + t) * DHh + d;
  qh[idx] = (_Float16)qv; kh[idx] = (_Float16)kv; vh[idx] = (_Float16)vv;
}

// i_pre / f_pre gate dots over concat(q,k,v) = 1536 features
__global__ __launch_bounds__(128)
void k_ifpre(const _Float16* __restrict__ qh, const _Float16* __restrict__ kh,
             const _Float16* __restrict__ vh,
             const float* __restrict__ Wi, const float* __restrict__ bi,
             const float* __restrict__ Wf, const float* __restrict__ bf,
             float* __restrict__ ip, float* __restrict__ fp) {
  int row = blockIdx.x, b = row / Csz, t = row % Csz;
  float pi[4] = {0, 0, 0, 0}, pf[4] = {0, 0, 0, 0};
  for (int c = threadIdx.x; c < 1536; c += 128) {
    int sec = c >> 9, ch = c & 511;
    int hd = ch >> 7, d = ch & 127;
    const _Float16* src = sec == 0 ? qh : (sec == 1 ? kh : vh);
    float v = (float)src[(((size_t)(b * NHh + hd)) * Tp + t) * DHh + d];
#pragma unroll
    for (int o = 0; o < 4; ++o) {
      pi[o] += v * Wi[o * 1536 + c];
      pf[o] += v * Wf[o * 1536 + c];
    }
  }
  __shared__ float sm[128];
  for (int o = 0; o < 8; ++o) {
    sm[threadIdx.x] = (o < 4) ? pi[o] : pf[o - 4];
    __syncthreads();
    for (int st = 64; st > 0; st >>= 1) {
      if (threadIdx.x < st) sm[threadIdx.x] += sm[threadIdx.x + st];
      __syncthreads();
    }
    if (threadIdx.x == 0) {
      int hh = o & 3;
      size_t oi = ((size_t)(b * NHh + hh)) * T_SEQ + t;
      if (o < 4) ip[oi] = sm[0] + bi[hh];
      else       fp[oi] = sm[0] + bf[hh];
    }
    __syncthreads();
  }
}

// per-(b,h) scan: csf = cumsum(logsigmoid(f)); g = i - csf; cm = cummax(g);
// m = csf + cm; emn = exp(-m)
__global__ void k_scan(const float* __restrict__ ip, const float* __restrict__ fp,
                       float* __restrict__ g, float* __restrict__ cm,
                       float* __restrict__ emn) {
  int bh = blockIdx.x * blockDim.x + threadIdx.x;
  if (bh >= Bsz * NHh) return;
  size_t base = (size_t)bh * T_SEQ;
  float csf = 0.f, cmv = -1e30f;
  for (int t = 0; t < T_SEQ; ++t) {
    float f  = fp[base + t];
    float lf = (f >= 0.f) ? -log1pf(__expf(-f)) : (f - log1pf(__expf(f)));
    csf += lf;
    float gi = ip[base + t] - csf;
    cmv = fmaxf(cmv, gi);
    g[base + t]   = gi;
    cm[base + t]  = cmv;
    emn[base + t] = __expf(-(csf + cmv));
  }
}

// ---------------- flash-style mLSTM attention ----------------
// grid (14 s-tiles, 64 bh); block 128 thr = 4 waves; wave owns 16 rows.
__global__ __launch_bounds__(128)
void k_attn(const _Float16* __restrict__ qh, const _Float16* __restrict__ kh,
            const _Float16* __restrict__ vh,
            const float* __restrict__ g, const float* __restrict__ cm,
            const float* __restrict__ emn, float* __restrict__ hatt) {
  __shared__ _Float16 Qs[64 * 128];
  __shared__ _Float16 Ks[64 * 128];
  __shared__ _Float16 VT[128 * 64];
  __shared__ _Float16 Ps[64 * 64];
  __shared__ float rs[64], cmr[64], emr[64];

  const int tid = threadIdx.x, lane = tid & 31, wave = tid >> 5;
  const int s0 = blockIdx.x * 64;
  const int bh = blockIdx.y, b = bh >> 2, hh = bh & 3;
  const size_t qbase = (size_t)bh * Tp * DHh;

  for (int idx = tid; idx < 64 * 16; idx += 128) {
    int r = idx >> 4, c = (idx & 15) * 8;
    *(v8h*)&Qs[r * 128 + c] = *(const v8h*)&qh[qbase + (size_t)(s0 + r) * 128 + c];
  }
  if (tid < 64) {
    int sg = s0 + tid;
    cmr[tid] = (sg < T_SEQ) ? cm[(size_t)bh * T_SEQ + sg] : 0.f;
    emr[tid] = (sg < T_SEQ) ? emn[(size_t)bh * T_SEQ + sg] : 0.f;
    rs[tid]  = 0.f;
  }
  v8f zf = {};
  v8f acc2[8];
#pragma unroll
  for (int i = 0; i < 8; ++i) acc2[i] = zf;
  const int ml = wave * 16;
  const float scale = 0.08838834764831845f;  // 1/sqrt(128)

  for (int jt = 0; jt <= blockIdx.x; ++jt) {
    int t0 = jt * 64;
    __syncthreads();  // previous Ks/VT/Ps consumers done
    for (int idx = tid; idx < 64 * 16; idx += 128) {
      int r = idx >> 4, c = (idx & 15) * 8;
      *(v8h*)&Ks[r * 128 + c] = *(const v8h*)&kh[qbase + (size_t)(t0 + r) * 128 + c];
    }
    for (int idx = tid; idx < 64 * 128; idx += 128) {
      int tt = idx >> 7, d = idx & 127;
      VT[d * 64 + tt] = vh[qbase + (size_t)(t0 + tt) * 128 + d];
    }
    __syncthreads();

    // S = Q @ K^T   (wave: 16 rows x 64 cols, K=128)
    v8f s[4];
#pragma unroll
    for (int i = 0; i < 4; ++i) s[i] = zf;
#pragma unroll
    for (int kk = 0; kk < 4; ++kk) {
      v16h fa = load_a(Qs, 128, ml, kk * 32, lane);
#pragma unroll
      for (int nt = 0; nt < 4; ++nt) {
        v16h fb = load_b(Ks, 128, nt * 16, kk * 32, lane);
        s[nt] = wmma_f16(fa, fb, s[nt]);
      }
    }
    // P = S * scale * exp(g[t]-cm[s]) with causal mask; accumulate row sums
#pragma unroll
    for (int nt = 0; nt < 4; ++nt) {
      int nn = nt * 16 + (lane & 15);
      int tg = t0 + nn;
#pragma unroll
      for (int r = 0; r < 8; ++r) {
        int sl = ml + 8 * (lane >> 4) + r;
        int sg = s0 + sl;
        float p = 0.f;
        if (tg <= sg && tg < T_SEQ && sg < T_SEQ)
          p = s[nt][r] * scale * __expf(g[(size_t)bh * T_SEQ + tg] - cmr[sl]);
        Ps[sl * 64 + nn] = (_Float16)p;
        atomicAdd(&rs[sl], p);
      }
    }
    __syncthreads();
    // acc2 += P @ V   (wave: 16 rows x 128 cols, K=64)
#pragma unroll
    for (int kk = 0; kk < 2; ++kk) {
      v16h fa = load_a(Ps, 64, ml, kk * 32, lane);
#pragma unroll
      for (int nt = 0; nt < 8; ++nt) {
        v16h fb = load_b(VT, 64, nt * 16, kk * 32, lane);
        acc2[nt] = wmma_f16(fa, fb, acc2[nt]);
      }
    }
  }
  __syncthreads();
#pragma unroll
  for (int nt = 0; nt < 8; ++nt) {
    int d = nt * 16 + (lane & 15);
#pragma unroll
    for (int r = 0; r < 8; ++r) {
      int sl = ml + 8 * (lane >> 4) + r;
      int sg = s0 + sl;
      if (sg < T_SEQ) {
        float norm = fmaxf(fabsf(rs[sl]), emr[sl]) + 1e-6f;
        hatt[((size_t)(b * T_SEQ + sg)) * INNER + hh * 128 + d] = acc2[nt][r] / norm;
      }
    }
  }
}

// per-head LN + (h + skip*xa) * silu(z) -> f16
__global__ __launch_bounds__(128)
void k_combine(const float* __restrict__ hatt, const float* __restrict__ xa,
               const float* __restrict__ xi, const float* __restrict__ mhlnw,
               const float* __restrict__ skip, _Float16* __restrict__ hzh) {
  int row = blockIdx.x, hd = blockIdx.y, d = threadIdx.x;
  float v = hatt[(size_t)row * INNER + hd * 128 + d];
  __shared__ float sm[128], sq[128];
  sm[d] = v; sq[d] = v * v; __syncthreads();
  for (int st = 64; st > 0; st >>= 1) {
    if (d < st) { sm[d] += sm[d + st]; sq[d] += sq[d + st]; }
    __syncthreads();
  }
  float m = sm[0] / 128.f, var = sq[0] / 128.f - m * m;
  float hn = (v - m) * rsqrtf(var + 1e-5f) * mhlnw[hd * 128 + d];
  int ch = hd * 128 + d;
  float z = xi[(size_t)row * 1024 + 512 + ch];
  float comb = (hn + skip[ch] * xa[(size_t)row * INNER + ch]) * silu_f(z);
  hzh[(size_t)row * INNER + ch] = (_Float16)comb;
}

// post LN + GELU -> f16
__global__ __launch_bounds__(256)
void k_postln(const float* __restrict__ x2, const float* __restrict__ plnw,
              _Float16* __restrict__ yh) {
  int row = blockIdx.x, n = threadIdx.x;
  float v = x2[(size_t)row * N2 + n];
  __shared__ float sm[256], sq[256];
  sm[n] = v; sq[n] = v * v; __syncthreads();
  for (int st = 128; st > 0; st >>= 1) {
    if (n < st) { sm[n] += sm[n + st]; sq[n] += sq[n + st]; }
    __syncthreads();
  }
  float m = sm[0] / (float)N2, var = sq[0] / (float)N2 - m * m;
  float ln = (v - m) * rsqrtf(var + 1e-5f) * plnw[n];
  yh[(size_t)row * N2 + n] = (_Float16)gelu_exact(ln);
}

// ---------------- host orchestration ----------------
extern "C" void kernel_launch(void* const* d_in, const int* in_sizes, int n_in,
                              void* d_out, int out_size, void* d_ws, size_t ws_size,
                              hipStream_t stream) {
  (void)in_sizes; (void)n_in; (void)out_size; (void)ws_size;
  const float* x      = (const float*)d_in[0];
  const float* Wseas  = (const float*)d_in[1];
  const float* bseas  = (const float*)d_in[2];
  const float* Wtrend = (const float*)d_in[3];
  const float* btrend = (const float*)d_in[4];
  const float* Wmm    = (const float*)d_in[5];
  const float* bmm    = (const float*)d_in[6];
  const float* bn_g   = (const float*)d_in[7];
  const float* bn_b   = (const float*)d_in[8];
  const float* ln1w   = (const float*)d_in[9];
  const float* Wup    = (const float*)d_in[10];
  const float* convw  = (const float*)d_in[11];
  const float* convb  = (const float*)d_in[12];
  const float* Wq     = (const float*)d_in[13];
  const float* Wk     = (const float*)d_in[14];
  const float* Wv     = (const float*)d_in[15];
  const float* Wi     = (const float*)d_in[16];
  const float* bi     = (const float*)d_in[17];
  const float* Wf     = (const float*)d_in[18];
  const float* bf     = (const float*)d_in[19];
  const float* mhlnw  = (const float*)d_in[20];
  const float* skip   = (const float*)d_in[21];
  const float* Wdown  = (const float*)d_in[22];
  const float* plnw   = (const float*)d_in[23];
  const float* Wmm2   = (const float*)d_in[24];
  const float* bmm2   = (const float*)d_in[25];
  float* out = (float*)d_out;

  char* ws = (char*)d_ws;
  size_t off = 0;
  auto alloc = [&](size_t bytes) -> void* {
    void* p = ws + off;
    off = (off + bytes + 255) & ~(size_t)255;
    return p;
  };
  float*     trend  = (float*)alloc((size_t)Mrows * Lsz * 4);
  _Float16*  A1     = (_Float16*)alloc((size_t)Mp * 672 * 2);
  _Float16*  W1     = (_Float16*)alloc((size_t)384 * 672 * 2);
  float*     bsum   = (float*)alloc(Lsz * 4);
  _Float16*  st_h   = (_Float16*)alloc((size_t)Mp * 352 * 2);
  _Float16*  Wmm_h  = (_Float16*)alloc((size_t)256 * 352 * 2);
  float*     hbuf   = (float*)alloc((size_t)Mrows * N2 * 4);
  float*     bnmu   = (float*)alloc(Csz * 4);
  float*     bnrstd = (float*)alloc(Csz * 4);
  float*     h0     = (float*)alloc((size_t)Mrows * N2 * 4);
  _Float16*  xn_h   = (_Float16*)alloc((size_t)Mp * 256 * 2);
  _Float16*  Wup_h  = (_Float16*)alloc((size_t)1024 * 256 * 2);
  float*     xi     = (float*)alloc((size_t)Mrows * 1024 * 4);
  float*     xa     = (float*)alloc((size_t)Mrows * INNER * 4);
  _Float16*  qh     = (_Float16*)alloc((size_t)64 * Tp * DHh * 2);
  _Float16*  kh     = (_Float16*)alloc((size_t)64 * Tp * DHh * 2);
  _Float16*  vh     = (_Float16*)alloc((size_t)64 * Tp * DHh * 2);
  float*     ip     = (float*)alloc((size_t)64 * T_SEQ * 4);
  float*     fpb    = (float*)alloc((size_t)64 * T_SEQ * 4);
  float*     gb     = (float*)alloc((size_t)64 * T_SEQ * 4);
  float*     cmb    = (float*)alloc((size_t)64 * T_SEQ * 4);
  float*     emnb   = (float*)alloc((size_t)64 * T_SEQ * 4);
  float*     hatt   = (float*)alloc((size_t)Mrows * INNER * 4);
  _Float16*  hz_h   = (_Float16*)alloc((size_t)Mp * INNER * 2);
  _Float16*  Wdn_h  = (_Float16*)alloc((size_t)256 * 512 * 2);
  float*     x2     = (float*)alloc((size_t)Mrows * N2 * 4);
  _Float16*  y_h    = (_Float16*)alloc((size_t)Mp * 256 * 2);
  _Float16*  Wm2_h  = (_Float16*)alloc((size_t)128 * 256 * 2);

  auto zfill = [&](_Float16* p, size_t n) {
    k_zero_h<<<dim3((unsigned)((n + 255) / 256)), 256, 0, stream>>>(p, n);
  };

  // stage 0: decomposition + packs
  k_trend<<<Mrows, 352, 0, stream>>>(x, trend);
  {
    size_t t = (size_t)Mp * 672;
    k_packA1<<<dim3((unsigned)((t + 255) / 256)), 256, 0, stream>>>(x, trend, A1);
  }
  {
    size_t t = (size_t)384 * 672;
    k_packW1<<<dim3((unsigned)((t + 255) / 256)), 256, 0, stream>>>(Wseas, Wtrend, W1);
  }
  k_bsum<<<2, 256, 0, stream>>>(bseas, btrend, bsum);
  auto cvt = [&](const float* s, _Float16* d, int N, int K, int Kp, int Nb) {
    size_t t = (size_t)Nb * Kp;
    k_cvtW<<<dim3((unsigned)((t + 255) / 256)), 256, 0, stream>>>(s, d, N, K, Kp, Nb);
  };
  cvt(Wmm,  Wmm_h, 256, 336, 352, 256);
  cvt(Wup,  Wup_h, 1024, 256, 256, 1024);
  cvt(Wdown, Wdn_h, 256, 512, 512, 256);
  cvt(Wmm2, Wm2_h, 96, 256, 256, 128);
  zfill(st_h, (size_t)Mp * 352);
  zfill(xn_h, (size_t)Mp * 256);
  zfill(qh, (size_t)64 * Tp * DHh);
  zfill(kh, (size_t)64 * Tp * DHh);
  zfill(vh, (size_t)64 * Tp * DHh);
  zfill(hz_h, (size_t)Mp * INNER);
  zfill(y_h, (size_t)Mp * 256);

  // G1: st = seasonal@Wseas^T + trend@Wtrend^T + (bseas+btrend)  -> f16
  k_gemm<<<dim3(6, 216), 128, 0, stream>>>(A1, W1, nullptr, st_h, bsum, nullptr,
                                           Mrows, 336, 672, 672, 672, 0, 352);
  // G2: h = st @ Wmm^T + bmm  -> f32
  k_gemm<<<dim3(4, 216), 128, 0, stream>>>(st_h, Wmm_h, hbuf, nullptr, bmm, nullptr,
                                           Mrows, 256, 352, 352, 352, 256, 0);
  // BN + GELU + LN1
  k_bnstats<<<Csz, 256, 0, stream>>>(hbuf, bnmu, bnrstd);
  k_ln1<<<Mrows, 256, 0, stream>>>(hbuf, bnmu, bnrstd, bn_g, bn_b, ln1w, h0, xn_h);
  // G3: xi = xn @ Wup^T
  k_gemm<<<dim3(16, 216), 128, 0, stream>>>(xn_h, Wup_h, xi, nullptr, nullptr, nullptr,
                                            Mrows, 1024, 256, 256, 256, 1024, 0);
  // conv + silu ; headwise q/k/v
  k_conv<<<Mrows, 512, 0, stream>>>(xi, convw, convb, xa);
  k_headwise<<<Mrows, 512, 0, stream>>>(xa, xi, Wq, Wk, Wv, qh, kh, vh);
  // gates + scan
  k_ifpre<<<Mrows, 128, 0, stream>>>(qh, kh, vh, Wi, bi, Wf, bf, ip, fpb);
  k_scan<<<2, 32, 0, stream>>>(ip, fpb, gb, cmb, emnb);
  // mLSTM attention
  k_attn<<<dim3(14, 64), 128, 0, stream>>>(qh, kh, vh, gb, cmb, emnb, hatt);
  // head LN + gate combine
  k_combine<<<dim3(Mrows, 4), 128, 0, stream>>>(hatt, xa, xi, mhlnw, skip, hz_h);
  // G4: x2 = h0 + hz @ Wdown^T
  k_gemm<<<dim3(4, 216), 128, 0, stream>>>(hz_h, Wdn_h, x2, nullptr, nullptr, h0,
                                           Mrows, 256, 512, 512, 512, 256, 0);
  // post LN + GELU
  k_postln<<<Mrows, 256, 0, stream>>>(x2, plnw, y_h);
  // G5: out = y @ Wmm2^T + bmm2
  k_gemm<<<dim3(2, 216), 128, 0, stream>>>(y_h, Wm2_h, out, nullptr, bmm2, nullptr,
                                           Mrows, PRED, 256, 256, 256, PRED, 0);
}
